// OuterProductMean_1898375545280
// MI455X (gfx1250) — matrix operand order repaired and checked by reading, not compiled, not run on hardware
//
#include <hip/hip_runtime.h>
#include <hip/hip_bf16.h>
#include <stdint.h>

// ---------------------------------------------------------------------------
// OuterProductMean for MI455X (gfx1250), bf16 WMMA path.
// B=1, S=128, N=256, C_IN=256, C_H=32, C_OUT=128.
// ---------------------------------------------------------------------------

typedef __bf16 bf16_t;
typedef bf16_t  v16bf  __attribute__((ext_vector_type(16)));
typedef float   v8f    __attribute__((ext_vector_type(8)));
typedef uint32_t u32x4 __attribute__((ext_vector_type(4)));

union Frag {
    u32x4 q[2];   // two 16-byte chunks per lane (8 VGPRs)
    v16bf v;      // 16 bf16 = WMMA 16x16x32 operand per lane
};

#define S_DIM 128
#define N_DIM 256
#define CIN   256
#define CH    32
#define COUT  128
#define KTOT  (CH * CH)   // 1024

// ---------------------------------------------------------------------------
// Kernel 1: LayerNorm + Wa/Wb projections + mask; emit bf16 panels
//   Abf[n][c][s], Bbf[n][d][s]  (s contiguous -> WMMA-friendly layout)
// One wave per (s,n) row. Lane l owns output channel l (CH==32==wave width).
// ---------------------------------------------------------------------------
__global__ __launch_bounds__(256) void ln_proj_kernel(
    const float* __restrict__ m, const float* __restrict__ mask,
    const float* __restrict__ gamma, const float* __restrict__ beta,
    const float* __restrict__ Wa, const float* __restrict__ Wb,
    bf16_t* __restrict__ Abf, bf16_t* __restrict__ Bbf)
{
    __shared__ float mn[8][CIN];          // 8 KB, one LN'd row per wave
    const int w = threadIdx.x >> 5;
    const int l = threadIdx.x & 31;
    const int row = blockIdx.x * 8 + w;   // row in [0, S*N)
    const int s = row >> 8;               // N = 256
    const int n = row & 255;

    const float* x = m + (size_t)row * CIN;
    float v[8];
    float sum = 0.f, sq = 0.f;
#pragma unroll
    for (int t = 0; t < 8; ++t) {
        float f = x[l * 8 + t];
        v[t] = f; sum += f; sq += f * f;
    }
#pragma unroll
    for (int off = 16; off > 0; off >>= 1) {   // wave32 reduction
        sum += __shfl_xor(sum, off, 32);
        sq  += __shfl_xor(sq,  off, 32);
    }
    const float mu   = sum * (1.0f / CIN);
    const float var  = sq  * (1.0f / CIN) - mu * mu;
    const float rstd = rsqrtf(var + 1e-5f);
#pragma unroll
    for (int t = 0; t < 8; ++t) {
        const int i = l * 8 + t;
        mn[w][i] = (v[t] - mu) * rstd * gamma[i] + beta[i];
    }
    __syncthreads();

    float aacc = 0.f, bacc = 0.f;
    const float* wa = Wa + (size_t)l * CIN;
    const float* wb = Wb + (size_t)l * CIN;
#pragma unroll 4
    for (int i = 0; i < CIN; ++i) {
        const float mv = mn[w][i];
        aacc = fmaf(mv, wa[i], aacc);
        bacc = fmaf(mv, wb[i], bacc);
    }
    const float mk = mask[row];
    Abf[((size_t)n * CH + l) * S_DIM + s] = (bf16_t)(aacc * mk);
    Bbf[((size_t)n * CH + l) * S_DIM + s] = (bf16_t)(bacc * mk);
}

// ---------------------------------------------------------------------------
// Kernel 2: Wo (f32, [COUT][KTOT]) -> bf16 (same layout; K contiguous)
// ---------------------------------------------------------------------------
__global__ __launch_bounds__(256) void wo_convert_kernel(
    const float* __restrict__ Wo, bf16_t* __restrict__ Wobf)
{
    const int idx = blockIdx.x * 256 + threadIdx.x;
    Wobf[idx] = (bf16_t)Wo[idx];
}

// ---------------------------------------------------------------------------
// Kernel 3: reciprocal pair-mask  rnm[i*N+j] = 1 / max(sum_s mask[s,i]*mask[s,j], 1)
// ---------------------------------------------------------------------------
__global__ __launch_bounds__(256) void rnm_kernel(
    const float* __restrict__ mask, float* __restrict__ rnm)
{
    const int idx = blockIdx.x * 256 + threadIdx.x;   // [0, N*N)
    const int i = idx >> 8, j = idx & 255;
    float s = 0.f;
#pragma unroll 4
    for (int t = 0; t < S_DIM; ++t)
        s = fmaf(mask[t * N_DIM + i], mask[t * N_DIM + j], s);
    rnm[idx] = 1.0f / fmaxf(s, 1.0f);
}

// ---------------------------------------------------------------------------
// Kernel 4: fused outer-product-mean + output projection, all on
// v_wmma_f32_16x16x32_bf16. Block = 4x4 (i,j) pairs, 8 waves (256 thr).
//  Stage 1: z_ij(32x32) = A_i(32x128) x B_j(128x32); 2 pairs/wave; z -> LDS bf16
//  Stage 2: Out(16 pairs x 128) = Z(16x1024) x Wo^T; wave w -> o-tile w
// ---------------------------------------------------------------------------
__global__ __launch_bounds__(256) void opm_wmma_kernel(
    const bf16_t* __restrict__ Abf, const bf16_t* __restrict__ Bbf,
    const bf16_t* __restrict__ Wobf, const float* __restrict__ rnm,
    const float* __restrict__ bo, float* __restrict__ out)
{
    __shared__ bf16_t Zl[16 * KTOT];   // 32 KB: 16 pairs x 1024 (c*32+d)
    const int w  = threadIdx.x >> 5;
    const int l  = threadIdx.x & 31;
    const int lo = l & 15;
    const int hi = l >> 4;             // lane half selects K sub-block
    const int i0 = blockIdx.y * 4;
    const int j0 = blockIdx.x * 4;

    // ---------------- stage 1 ----------------
#pragma unroll
    for (int pp = 0; pp < 2; ++pp) {
        const int p  = w * 2 + pp;          // pair 0..15
        const int pi = p >> 2, pj = p & 3;
        const bf16_t* Ap = Abf + (size_t)(i0 + pi) * CH * S_DIM;  // [c][s]
        const bf16_t* Bp = Bbf + (size_t)(j0 + pj) * CH * S_DIM;  // [d][s]

        v8f acc[2][2];
#pragma unroll
        for (int mt = 0; mt < 2; ++mt)
#pragma unroll
            for (int nt = 0; nt < 2; ++nt)
                acc[mt][nt] = (v8f){0.f,0.f,0.f,0.f,0.f,0.f,0.f,0.f};

#pragma unroll
        for (int kk = 0; kk < 4; ++kk) {    // K = s, 4 steps of 32
            const int k0 = kk * 32;
            Frag af[2], bg[2];
#pragma unroll
            for (int mt = 0; mt < 2; ++mt) {
                // A 16x32 bf16: regs0-3 K=(hi?8:0)+0..7, regs4-7 = +16
                const bf16_t* base = Ap + (size_t)(mt * 16 + lo) * S_DIM + k0 + hi * 8;
                af[mt].q[0] = *(const u32x4*)(base);
                af[mt].q[1] = *(const u32x4*)(base + 16);
            }
#pragma unroll
            for (int nt = 0; nt < 2; ++nt) {
                // B 32x16 bf16: regs0-3 K=(hi?8:0)+0..7, regs4-7 = +16
                // -> same K order as A, from the [d][s] row
                const bf16_t* base = Bp + (size_t)(nt * 16 + lo) * S_DIM + k0 + hi * 8;
                bg[nt].q[0] = *(const u32x4*)(base);
                bg[nt].q[1] = *(const u32x4*)(base + 16);
            }
#pragma unroll
            for (int mt = 0; mt < 2; ++mt)
#pragma unroll
                for (int nt = 0; nt < 2; ++nt)
                    acc[mt][nt] = __builtin_amdgcn_wmma_f32_16x16x32_bf16(
                        false, af[mt].v, false, bg[nt].v,
                        (short)0, acc[mt][nt], false, false);
        }
        // z tile -> LDS (bf16), flattened k = c*32 + d
#pragma unroll
        for (int mt = 0; mt < 2; ++mt)
#pragma unroll
            for (int nt = 0; nt < 2; ++nt)
#pragma unroll
                for (int vv = 0; vv < 8; ++vv) {
                    const int c = mt * 16 + vv + hi * 8;   // C/D layout: M=v+8*hi
                    const int d = nt * 16 + lo;            // N = lane%16
                    Zl[p * KTOT + c * CH + d] = (bf16_t)acc[mt][nt][vv];
                }
    }
    __syncthreads();

    // ---------------- stage 2 ----------------
    v8f acc2 = (v8f){0.f,0.f,0.f,0.f,0.f,0.f,0.f,0.f};
    const bf16_t* Wrow = Wobf + (size_t)(w * 16 + lo) * KTOT;   // o = w*16+lo row
#pragma unroll 2
    for (int ks = 0; ks < 32; ++ks) {       // K = 1024, 32 steps
        const int k0 = ks * 32;
        Frag af, bg;
        const bf16_t* abase = &Zl[lo * KTOT + k0 + hi * 8];     // M = pair = lane%16
        af.q[0] = *(const u32x4*)(abase);
        af.q[1] = *(const u32x4*)(abase + 16);
        const bf16_t* bbase = Wrow + k0 + hi * 8;               // B = Wo^T
        bg.q[0] = *(const u32x4*)(bbase);
        bg.q[1] = *(const u32x4*)(bbase + 16);
        acc2 = __builtin_amdgcn_wmma_f32_16x16x32_bf16(
            false, af.v, false, bg.v, (short)0, acc2, false, false);
    }

    // epilogue: scale by 1/num_mask, add bias
    const int o = w * 16 + lo;
    const float bov = bo[o];
#pragma unroll
    for (int vv = 0; vv < 8; ++vv) {
        const int p = vv + hi * 8;                  // C layout: M = v + 8*hi
        const int i = i0 + (p >> 2);
        const int j = j0 + (p & 3);
        const int ij = i * N_DIM + j;
        out[(size_t)ij * COUT + o] = acc2[vv] * rnm[ij] + bov;
    }
}

// ---------------------------------------------------------------------------
// Host launcher
// ---------------------------------------------------------------------------
extern "C" void kernel_launch(void* const* d_in, const int* in_sizes, int n_in,
                              void* d_out, int out_size, void* d_ws, size_t ws_size,
                              hipStream_t stream) {
    const float* m     = (const float*)d_in[0];
    const float* mask  = (const float*)d_in[1];
    const float* gamma = (const float*)d_in[2];
    const float* beta  = (const float*)d_in[3];
    const float* Wa    = (const float*)d_in[4];
    const float* Wb    = (const float*)d_in[5];
    const float* Wo    = (const float*)d_in[6];
    const float* bo    = (const float*)d_in[7];
    float* out = (float*)d_out;

    char* ws = (char*)d_ws;
    bf16_t* Abf  = (bf16_t*)(ws);                               // 256*32*128*2 = 2 MB
    bf16_t* Bbf  = (bf16_t*)(ws + (2u << 20));                  // 2 MB
    bf16_t* Wobf = (bf16_t*)(ws + (4u << 20));                  // 128*1024*2 = 256 KB
    float*  rnm  = (float*) (ws + (4u << 20) + (256u << 10));   // 256*256*4 = 256 KB

    // 1) LN + projections -> bf16 panels (S*N = 32768 rows, 8 rows/block)
    ln_proj_kernel<<<dim3(S_DIM * N_DIM / 8), dim3(256), 0, stream>>>(
        m, mask, gamma, beta, Wa, Wb, Abf, Bbf);
    // 2) Wo -> bf16 (128*1024 = 131072 elems)
    wo_convert_kernel<<<dim3(COUT * KTOT / 256), dim3(256), 0, stream>>>(Wo, Wobf);
    // 3) reciprocal pair-mask (256*256 elems)
    rnm_kernel<<<dim3(N_DIM * N_DIM / 256), dim3(256), 0, stream>>>(mask, rnm);
    // 4) fused outer-product-mean + projection (64x64 blocks of 4x4 pairs)
    opm_wmma_kernel<<<dim3(N_DIM / 4, N_DIM / 4), dim3(256), 0, stream>>>(
        Abf, Bbf, Wobf, rnm, bo, out);
}
